// QKVAttentionLegacy_63917703299422
// MI455X (gfx1250) — matrix operand order repaired and checked by reading, not compiled, not run on hardware
//
#include <hip/hip_runtime.h>

// QKV attention (flash-style) for MI455X / gfx1250, wave32 + WMMA f16.
// qkv: (2, 3072, 2048) f32, layout qkv[b][c][t]; n_heads=16 -> D=64, L=2048.
// out: (2, 1024, 2048) f32.
//
// S^T = K x Q (rows=s, cols=t): each lane owns one t-column -> softmax is a
// per-lane register reduction + one xor-16 shuffle. O^T = V x P^T with V's
// natural layout as the A matrix; P^T passes register-to-register (half-wave
// swap only). Softmax runs in the log2 domain (log2e folded into the Q scale)
// so every exponent is a single native v_exp_f32.

typedef __attribute__((ext_vector_type(16))) _Float16 v16h;
typedef __attribute__((ext_vector_type(8)))  _Float16 v8h;
typedef __attribute__((ext_vector_type(4)))  _Float16 v4h;
typedef __attribute__((ext_vector_type(8)))  float    v8f;
typedef __attribute__((ext_vector_type(4)))  int      v4i;

#define L_SEQ   2048
#define D_HEAD  64
#define N_HEADS 16
#define WIDTH   3072
#define BLOCK_T 128             // 8 waves x 16 query columns
#define BLOCK_S 64              // key/value block
#define NBLK    (L_SEQ / BLOCK_S)
#define Q_SCALE 0.180336880111f // (1/8) * log2(e): softmax in log2 domain

static __device__ __forceinline__ v16h cat8(v8h lo, v8h hi) {
  return __builtin_shufflevector(lo, hi, 0,1,2,3,4,5,6,7,8,9,10,11,12,13,14,15);
}
static __device__ __forceinline__ v8h ldsh8(const _Float16* p) {
  return *reinterpret_cast<const v8h*>(p);  // 16B-aligned -> ds_load_b128
}
static __device__ __forceinline__ float fcomp(const float4& v, int i) {
  return (&v.x)[i];
}

__global__ __launch_bounds__(256) void qkv_attn_kernel(const float* __restrict__ qkv,
                                                       float* __restrict__ out) {
  // LDS (32 KB): Q 128x64 f16 [t][c] | K 64x64 f16 [s][c] | V 64x64 f16 [c][s]
  __shared__ __align__(16) _Float16 smem[16384];
  _Float16* sQ = smem;           // [t][c]
  _Float16* sK = smem + 8192;    // [s][c]
  _Float16* sV = smem + 12288;   // [c][s]

  const int tid  = threadIdx.x;
  const int wave = tid >> 5;
  const int lane = tid & 31;
  const int lh   = lane >> 4;    // lane half (fragment K-group select)
  const int lm   = lane & 15;

  const int t0 = blockIdx.x * BLOCK_T;   // query tile base
  const int bh = blockIdx.y;             // flattened (b, head)
  const int b  = bh >> 4, h = bh & 15;

  const float* Qg = qkv + (size_t)(b * WIDTH + h * 3 * D_HEAD) * L_SEQ;
  const float* Kg = Qg + (size_t)D_HEAD * L_SEQ;
  const float* Vg = Kg + (size_t)D_HEAD * L_SEQ;

  // ---- Stage Q tile: [c][t] -> [t][c], fold scale*log2e, packed b64 stores
  {
    const int t4 = (tid & 31) << 2;                 // 0..124
    #pragma unroll
    for (int r = 0; r < 2; ++r) {
      const int c0 = (((tid >> 5) + r * 8) << 2);   // 0..60
      float4 q[4];
      #pragma unroll
      for (int i = 0; i < 4; ++i)
        q[i] = *reinterpret_cast<const float4*>(Qg + (size_t)(c0 + i) * L_SEQ + t0 + t4);
      #pragma unroll
      for (int i = 0; i < 4; ++i) {
        v4h hv = { (_Float16)(Q_SCALE * fcomp(q[0], i)), (_Float16)(Q_SCALE * fcomp(q[1], i)),
                   (_Float16)(Q_SCALE * fcomp(q[2], i)), (_Float16)(Q_SCALE * fcomp(q[3], i)) };
        *reinterpret_cast<v4h*>(&sQ[(t4 + i) * 64 + c0]) = hv;
      }
    }
  }

  // ---- Pipelined K/V staging: 4x4 sub-tile per thread, one block ahead
  const int kc0 = (tid >> 4) << 2;   // 0..60 (channel group)
  const int ks4 = (tid & 15) << 2;   // 0..60 (s group)
  float4 kr[4], vr[4];
  #pragma unroll
  for (int i = 0; i < 4; ++i) {
    kr[i] = *reinterpret_cast<const float4*>(Kg + (size_t)(kc0 + i) * L_SEQ + ks4);
    vr[i] = *reinterpret_cast<const float4*>(Vg + (size_t)(kc0 + i) * L_SEQ + ks4);
  }
  __syncthreads();  // sQ visible

  // Q as B-fragments (col = t = lane&15, 16 contiguous K=c per lane)
  v16h bq[2];
  #pragma unroll
  for (int kc = 0; kc < 2; ++kc) {
    const int qb = (wave * 16 + lm) * 64 + kc * 32 + lh * 16;
    bq[kc] = cat8(ldsh8(&sQ[qb]), ldsh8(&sQ[qb + 8]));
  }

  const v8f zero8 = {0.f,0.f,0.f,0.f,0.f,0.f,0.f,0.f};
  v8f o[4];                       // O^T: rows c (4 tiles), col t = this lane
  #pragma unroll
  for (int j = 0; j < 4; ++j) o[j] = zero8;
  float m = -3.0e38f, l = 0.0f;   // per-lane running stats (log2 domain)

  for (int sb = 0; sb < NBLK; ++sb) {
    // ---- Commit staged K (transposed [s][c]) and V (natural [c][s]) to LDS
    #pragma unroll
    for (int i = 0; i < 4; ++i) {
      v4h kh = { (_Float16)fcomp(kr[0], i), (_Float16)fcomp(kr[1], i),
                 (_Float16)fcomp(kr[2], i), (_Float16)fcomp(kr[3], i) };
      *reinterpret_cast<v4h*>(&sK[(ks4 + i) * 64 + kc0]) = kh;
      v4h vh = { (_Float16)fcomp(vr[i], 0), (_Float16)fcomp(vr[i], 1),
                 (_Float16)fcomp(vr[i], 2), (_Float16)fcomp(vr[i], 3) };
      *reinterpret_cast<v4h*>(&sV[(kc0 + i) * 64 + ks4]) = vh;
    }
    __syncthreads();

    // ---- Issue next block's global loads (latency hidden behind WMMAs)
    if (sb + 1 < NBLK) {
      const int ns = (sb + 1) * BLOCK_S + ks4;
      #pragma unroll
      for (int i = 0; i < 4; ++i) {
        kr[i] = *reinterpret_cast<const float4*>(Kg + (size_t)(kc0 + i) * L_SEQ + ns);
        vr[i] = *reinterpret_cast<const float4*>(Vg + (size_t)(kc0 + i) * L_SEQ + ns);
      }
    }

    // ---- S^T = Kblock x Qtile : rows s (4 tiles), cols t; f32 accum
    v8f acc[4];
    #pragma unroll
    for (int j = 0; j < 4; ++j) {
      acc[j] = zero8;
      #pragma unroll
      for (int kc = 0; kc < 2; ++kc) {
        const int ab = (j * 16 + lm) * 64 + kc * 32 + lh * 16;  // A row = s, K = c
        v16h ak = cat8(ldsh8(&sK[ab]), ldsh8(&sK[ab + 8]));
        acc[j] = __builtin_amdgcn_wmma_f32_16x16x32_f16(
            false, ak, false, bq[kc], (short)0, acc[j], false, false);
      }
    }

    // ---- Online softmax (log2 domain): per-lane reduce + one xor-16 combine
    float rm = acc[0][0];
    #pragma unroll
    for (int j = 0; j < 4; ++j)
      #pragma unroll
      for (int g = 0; g < 8; ++g) rm = fmaxf(rm, acc[j][g]);
    rm = fmaxf(rm, __shfl_xor(rm, 16, 32));
    const float mn    = fmaxf(m, rm);
    const float alpha = __builtin_amdgcn_exp2f(m - mn);   // native v_exp_f32
    float rs = 0.0f;
    #pragma unroll
    for (int j = 0; j < 4; ++j)
      #pragma unroll
      for (int g = 0; g < 8; ++g) {
        float p = __builtin_amdgcn_exp2f(acc[j][g] - mn);
        acc[j][g] = p;
        rs += p;
      }
    rs += __shfl_xor(rs, 16, 32);
    l = l * alpha + rs;
    m = mn;
    #pragma unroll
    for (int j = 0; j < 4; ++j)
      #pragma unroll
      for (int g = 0; g < 8; ++g) o[j][g] *= alpha;

    // ---- O^T += Vblock x P^T. P^T is per-lane column-matched to the B
    //      layout; only the half-wave s-rows need exchanging (xor-16).
    #pragma unroll
    for (int kc = 0; kc < 2; ++kc) {
      v8h pa, pb;
      #pragma unroll
      for (int g = 0; g < 8; ++g) {
        pa[g] = (_Float16)acc[2 * kc][g];
        pb[g] = (_Float16)acc[2 * kc + 1][g];
      }
      v4i ia = __builtin_bit_cast(v4i, pa), ib = __builtin_bit_cast(v4i, pb);
      v4i sa, sbv;
      #pragma unroll
      for (int i = 0; i < 4; ++i) {
        sa[i]  = __shfl_xor(ia[i], 16, 32);
        sbv[i] = __shfl_xor(ib[i], 16, 32);
      }
      v8h sah = __builtin_bit_cast(v8h, sa), sbh = __builtin_bit_cast(v8h, sbv);
      v8h lo = lh ? sbh : pa;   // K rows: lanes<16 -> s 0..15 ; lanes>=16 -> s 16..31
      v8h hi = lh ? pb  : sah;
      v16h bp = cat8(lo, hi);
      #pragma unroll
      for (int jc = 0; jc < 4; ++jc) {
        const int vb = (jc * 16 + lm) * 64 + kc * 32 + lh * 16;  // A = V[c][s], natural
        v16h av = cat8(ldsh8(&sV[vb]), ldsh8(&sV[vb + 8]));
        o[jc] = __builtin_amdgcn_wmma_f32_16x16x32_f16(
            false, av, false, bp, (short)0, o[jc], false, false);
      }
    }
    __syncthreads();  // compute done before next staging overwrites sK/sV
  }

  // ---- Epilogue: O^T /= l, write out[b][h*64+c][t] (f32, streaming stores)
  const float inv = 1.0f / l;
  const int   tg  = t0 + wave * 16 + lm;
  #pragma unroll
  for (int jc = 0; jc < 4; ++jc) {
    #pragma unroll
    for (int g = 0; g < 8; ++g) {
      const int c = jc * 16 + g + lh * 8;
      float* p = out + (size_t)(b * (N_HEADS * D_HEAD) + h * D_HEAD + c) * L_SEQ + tg;
      __builtin_nontemporal_store(o[jc][g] * inv, p);
    }
  }
}

extern "C" void kernel_launch(void* const* d_in, const int* in_sizes, int n_in,
                              void* d_out, int out_size, void* d_ws, size_t ws_size,
                              hipStream_t stream) {
  (void)in_sizes; (void)n_in; (void)out_size; (void)d_ws; (void)ws_size;
  const float* qkv = (const float*)d_in[0];   // (2, 3072, 2048) f32
  float* out = (float*)d_out;                 // (2, 1024, 2048) f32
  dim3 grid(L_SEQ / BLOCK_T, 32);             // 16 query tiles x 32 (b,head)
  qkv_attn_kernel<<<grid, dim3(256), 0, stream>>>(qkv, out);
}
